// Encoding_76888504533767
// MI455X (gfx1250) — matrix operand order repaired and checked by reading, not compiled, not run on hardware
//
#include <hip/hip_runtime.h>

// Problem constants (match reference)
#define B_   64
#define N_   4096
#define D_   256
#define K_   32

// Tiling
#define ROWS     128                      // rows of x per LDS tile
#define WG_PER_B 8                        // partitions of N per batch
#define TILES    (N_ / (ROWS * WG_PER_B)) // 4
#define THREADS  256                      // 8 wave32 waves
#define XPAD     264                      // f16 elems per padded xs row (bank spread)
#define CPAD     264
#define WROWPAD  136                      // padded row length of transposed weights

typedef __attribute__((ext_vector_type(16))) _Float16 v16h;
typedef __attribute__((ext_vector_type(8)))  _Float16 v8h;
typedef __attribute__((ext_vector_type(4)))  _Float16 v4h;
typedef __attribute__((ext_vector_type(8)))  float    v8f;

typedef unsigned int u32x4 __attribute__((ext_vector_type(4)));
typedef int          i32x8 __attribute__((ext_vector_type(8)));
typedef int          i32x4 __attribute__((ext_vector_type(4)));

__device__ __forceinline__ v16h cat8(v8h lo, v8h hi) {
  return __builtin_shufflevector(lo, hi, 0,1,2,3,4,5,6,7,8,9,10,11,12,13,14,15);
}

__global__ void zero_out_kernel(float* __restrict__ out, int n) {
  int i = blockIdx.x * blockDim.x + threadIdx.x;
  if (i < n) out[i] = 0.0f;
}

__global__ __launch_bounds__(THREADS, 1)
void deepten_kernel(const float* __restrict__ xg, const float* __restrict__ Cg,
                    const float* __restrict__ Sg, float* __restrict__ out) {
  __shared__ alignas(16) _Float16 xs[ROWS][XPAD];    // x tile, f16
  __shared__ alignas(16) _Float16 Ch[K_][CPAD];      // codebook, f16
  __shared__ alignas(16) _Float16 wsT[K_][WROWPAD];  // softmax weights, transposed
  __shared__ alignas(16) float Craw[K_][D_];         // codebook f32 (TDM destination)
  __shared__ float x2s[ROWS];
  __shared__ float c2s[K_];
  __shared__ float Ss[K_];
  __shared__ float wsum[K_];

  const int t    = threadIdx.x;
  const int lane = t & 31;
  const int wv   = t >> 5;                 // wave id 0..7
  const int b    = blockIdx.x / WG_PER_B;
  const int part = blockIdx.x % WG_PER_B;

  const int kk = lane & 15;                // column-in-tile (WMMA B/C lane layout)
  const int lh = lane >> 4;                // half-wave select

  // ---- TDM: DMA the codebook (2D tensor 256x32 f32, whole-tensor tile) to LDS ----
  if (t == 0) {
    unsigned long long ga = (unsigned long long)(const void*)Cg;
    // flat LDS addresses map to LDS offsets via addr[31:0] (aperture rule)
    unsigned lds = (unsigned)(unsigned long long)(const void*)&Craw[0][0];
    u32x4 g0;
    g0[0] = 1u;                                                // count=1, user D#
    g0[1] = lds;                                               // lds_addr
    g0[2] = (unsigned)ga;                                      // global_addr[31:0]
    g0[3] = (unsigned)((ga >> 32) & 0x1FFFFFFu) | (2u << 30);  // addr[56:32] | type=2
    i32x8 g1;
    g1[0] = (int)(2u << 16);    // workgroup_mask=0, data_size=2 (4 bytes)
    g1[1] = (int)(256u << 16);  // tensor_dim0 = 256 (low 16 in bits[31:16])
    g1[2] = (int)(32u << 16);   // tensor_dim0 hi=0 | tensor_dim1 = 32
    g1[3] = (int)(256u << 16);  // tensor_dim1 hi=0 | tile_dim0 = 256
    g1[4] = 32;                 // tile_dim1 = 32 | tile_dim2 = 0
    g1[5] = 256;                // tensor_dim0_stride = 256 (low 32)
    g1[6] = 0;                  // stride hi / tensor_dim1_stride lo
    g1[7] = 0;
    i32x4 z4 = {0, 0, 0, 0};
    i32x8 z8 = {0, 0, 0, 0, 0, 0, 0, 0};
    __builtin_amdgcn_tensor_load_to_lds(g0, g1, z4, z4, z8, 0);
  }
  __builtin_amdgcn_s_wait_tensorcnt(0);
  if (t < K_) { Ss[t] = Sg[t]; wsum[t] = 0.0f; }
  __syncthreads();

  // ---- setup from LDS copy: convert C to f16, compute ||C_k||^2 ----
  #pragma unroll
  for (int i = 0; i < (K_ * D_) / THREADS; ++i) {
    int idx = t + i * THREADS;
    Ch[idx >> 8][idx & 255] = (_Float16)Craw[idx >> 8][idx & 255];
  }
  if (t < K_) {
    float s = 0.0f;
    #pragma unroll 8
    for (int d = 0; d < D_; ++d) { float c = Craw[t][d]; s += c * c; }
    c2s[t] = s;
  }

  // persistent E accumulators: wave owns d-tiles {2wv, 2wv+1} x k-tiles {0,1}
  v8f e00 = {}, e01 = {}, e10 = {}, e11 = {};
  float lws0 = 0.0f, lws1 = 0.0f;          // local sum of weights for k=kk, k=kk+16
  const int dt0 = wv * 2, dt1 = wv * 2 + 1;
  const int d0 = dt0 * 16 + kk, d1 = dt1 * 16 + kk;

  __syncthreads();

  for (int tile = 0; tile < TILES; ++tile) {
    const int row0 = (part * TILES + tile) * ROWS;
    const float* xt = xg + ((size_t)b * N_ + row0) * (size_t)D_;

    // prefetch next tile into cache hierarchy (global_prefetch_b8)
    if (tile + 1 < TILES) {
      const char* np = (const char*)(xt + (size_t)ROWS * D_);
      #pragma unroll
      for (int i = 0; i < 4; ++i) __builtin_prefetch(np + t * 512 + i * 128, 0, 1);
    }

    // ---- load x tile (fully coalesced b128) and convert to f16 in LDS ----
    #pragma unroll
    for (int i = 0; i < (ROWS * D_ / 4) / THREADS; ++i) {   // 32 float4 per thread
      int f  = t + i * THREADS;
      int r  = f >> 6;        // 64 float4 per row
      int c4 = f & 63;
      float4 v = ((const float4*)xt)[f];
      v4h h;
      h[0] = (_Float16)v.x; h[1] = (_Float16)v.y;
      h[2] = (_Float16)v.z; h[3] = (_Float16)v.w;
      *(v4h*)&xs[r][c4 * 4] = h;
    }
    __syncthreads();

    // ---- ||x_row||^2 from LDS: 2 threads per row, shfl-pair combine ----
    {
      int r = t >> 1, hf = t & 1;
      float s = 0.0f;
      #pragma unroll
      for (int j = 0; j < 16; ++j) {
        v8h v = *(const v8h*)&xs[r][hf * 128 + j * 8];
        #pragma unroll
        for (int q = 0; q < 8; ++q) { float f = (float)v[q]; s += f * f; }
      }
      s += __shfl_xor(s, 1);
      if (hf == 0) x2s[r] = s;
    }
    __syncthreads();

    // ---- GEMM1: xc[16 rows x 32 k] for this wave's row block ----
    v8f acc0 = {}, acc1 = {};
    const int r0   = wv * 16;
    const int arow = r0 + kk;              // A layout: M = lane%16
    #pragma unroll
    for (int ks = 0; ks < D_ / 32; ++ks) { // 8 K-steps
      int dbA = ks * 32 + lh * 8;          // A: halves 0-7 -> K=dbA.., 8-15 -> K=dbA+16..
      v16h a  = cat8(*(const v8h*)&xs[arow][dbA], *(const v8h*)&xs[arow][dbA + 16]);
      int dbB = ks * 32 + lh * 8;          // B: halves 0-7 -> K=dbB.., 8-15 -> K=dbB+16.., N=lane%16
      v16h b0 = cat8(*(const v8h*)&Ch[kk][dbB],      *(const v8h*)&Ch[kk][dbB + 16]);
      v16h b1 = cat8(*(const v8h*)&Ch[kk + 16][dbB], *(const v8h*)&Ch[kk + 16][dbB + 16]);
      acc0 = __builtin_amdgcn_wmma_f32_16x16x32_f16(false, a, false, b0, (short)0, acc0, false, false);
      acc1 = __builtin_amdgcn_wmma_f32_16x16x32_f16(false, a, false, b1, (short)0, acc1, false, false);
    }

    // ---- softmax over k=32 per row (rows live across 16-lane halves) ----
    #pragma unroll
    for (int j = 0; j < 8; ++j) {
      int row  = r0 + j + 8 * lh;          // C layout: M = j + 8*(lane/16)
      float x2v = x2s[row];
      float l0 = -Ss[kk]      * (x2v - 2.0f * acc0[j] + c2s[kk]);
      float l1 = -Ss[kk + 16] * (x2v - 2.0f * acc1[j] + c2s[kk + 16]);
      float m = fmaxf(l0, l1);
      m = fmaxf(m, __shfl_xor(m, 1));
      m = fmaxf(m, __shfl_xor(m, 2));
      m = fmaxf(m, __shfl_xor(m, 4));
      m = fmaxf(m, __shfl_xor(m, 8));
      float e0 = __expf(l0 - m), e1 = __expf(l1 - m);
      float sm = e0 + e1;
      sm += __shfl_xor(sm, 1);
      sm += __shfl_xor(sm, 2);
      sm += __shfl_xor(sm, 4);
      sm += __shfl_xor(sm, 8);
      float inv = 1.0f / sm;
      float w0 = e0 * inv, w1 = e1 * inv;
      wsT[kk][row]      = (_Float16)w0;    // transposed store -> vector A-frag loads
      wsT[kk + 16][row] = (_Float16)w1;
      lws0 += w0;
      lws1 += w1;
    }
    __syncthreads();

    // ---- GEMM2: E_partial[32 k x 256 d] += w^T * x ----
    #pragma unroll
    for (int ks = 0; ks < ROWS / 32; ++ks) {  // 4 K-steps over rows
      int nb = ks * 32;
      int na = nb + lh * 16;                  // A: M=k per lane, halves sweep n
      v16h a0 = cat8(*(const v8h*)&wsT[kk][na],      *(const v8h*)&wsT[kk][na + 8]);
      v16h a1 = cat8(*(const v8h*)&wsT[kk + 16][na], *(const v8h*)&wsT[kk + 16][na + 8]);
      v16h b0, b1;
      #pragma unroll
      for (int h = 0; h < 16; ++h) {          // B: N=d fixed per lane, halves sweep n
        int n = nb + lh * 16 + h;
        b0[h] = xs[n][d0];
        b1[h] = xs[n][d1];
      }
      e00 = __builtin_amdgcn_wmma_f32_16x16x32_f16(false, a0, false, b0, (short)0, e00, false, false);
      e01 = __builtin_amdgcn_wmma_f32_16x16x32_f16(false, a0, false, b1, (short)0, e01, false, false);
      e10 = __builtin_amdgcn_wmma_f32_16x16x32_f16(false, a1, false, b0, (short)0, e10, false, false);
      e11 = __builtin_amdgcn_wmma_f32_16x16x32_f16(false, a1, false, b1, (short)0, e11, false, false);
    }
    __syncthreads();   // before next tile overwrites xs/wsT
  }

  // ---- reduce sum-of-weights into LDS ----
  lws0 += __shfl_xor(lws0, 16);
  lws1 += __shfl_xor(lws1, 16);
  if (lane < 16) {
    atomicAdd(&wsum[kk], lws0);        // ds_add_f32
    atomicAdd(&wsum[kk + 16], lws1);
  }
  __syncthreads();

  // ---- writeback: E_partial - wsum_partial * C, atomically into global ----
  float* ob = out + (size_t)b * K_ * D_;
  #pragma unroll
  for (int j = 0; j < 8; ++j) {
    int k0 = j + 8 * lh;               // k-tile 0
    int k1 = 16 + k0;                  // k-tile 1
    float ws0 = wsum[k0], ws1 = wsum[k1];
    unsafeAtomicAdd(&ob[k0 * D_ + d0], e00[j] - ws0 * Craw[k0][d0]);
    unsafeAtomicAdd(&ob[k0 * D_ + d1], e01[j] - ws0 * Craw[k0][d1]);
    unsafeAtomicAdd(&ob[k1 * D_ + d0], e10[j] - ws1 * Craw[k1][d0]);
    unsafeAtomicAdd(&ob[k1 * D_ + d1], e11[j] - ws1 * Craw[k1][d1]);
  }
}

extern "C" void kernel_launch(void* const* d_in, const int* in_sizes, int n_in,
                              void* d_out, int out_size, void* d_ws, size_t ws_size,
                              hipStream_t stream) {
  const float* x = (const float*)d_in[0];
  const float* C = (const float*)d_in[1];
  const float* S = (const float*)d_in[2];
  float* out = (float*)d_out;

  // d_out is poisoned by harness; accumulation via atomics requires zeroing.
  zero_out_kernel<<<(out_size + 255) / 256, 256, 0, stream>>>(out, out_size);

  dim3 grid(B_ * WG_PER_B);
  dim3 block(THREADS);
  deepten_kernel<<<grid, block, 0, stream>>>(x, C, S, out);
}
